// WGQAAttention_27410481283079
// MI455X (gfx1250) — compile-verified, matrix operand order checked by reading, not run on hardware
//
#include <hip/hip_runtime.h>
#include <hip/hip_bf16.h>
#include <math.h>

// ---------- constants ----------
#define D_MODEL 2048
#define N_HEADS 16
#define KV_HEADS 4
#define HEAD_DIM 128
#define SEQ_T   2048
#define BATCH   2
#define M_ROWS  (BATCH * SEQ_T)       // 4096
#define KV_DIM  (KV_HEADS * HEAD_DIM) // 512

// ---------- WMMA vector types ----------
typedef __attribute__((ext_vector_type(16))) __bf16 v16bf;
typedef __attribute__((ext_vector_type(8)))  float  v8f;
typedef int v4i_vs __attribute__((vector_size(16)));

struct alignas(16) U128 { unsigned int w[4]; };
union FragU { v16bf v; U128 q[2]; };

#if defined(__HIP_DEVICE_COMPILE__) && __has_builtin(__builtin_amdgcn_global_load_async_to_lds_b128)
#define USE_ASYNC_LDS 1
typedef __attribute__((address_space(1))) v4i_vs* gv4i_p;
typedef __attribute__((address_space(3))) v4i_vs* lv4i_p;
#else
#define USE_ASYNC_LDS 0
#endif

__device__ __forceinline__ unsigned short f2bf(float f) {
    unsigned int u = __float_as_uint(f);
    unsigned int r = u + 0x7FFFu + ((u >> 16) & 1u);   // RNE
    return (unsigned short)(r >> 16);
}

// Load a 16x32 bf16 fragment (A-layout / transposed-B layout) from a
// row-major matrix with leading dimension ld (elements). Per ISA 7.12.2:
// lane<16 : row=lane,    K = {0..7, 16..23}
// lane>=16: row=lane-16, K = {8..15, 24..31}
__device__ __forceinline__ v16bf load_frag(const unsigned short* __restrict__ base, int ld) {
    const int lane = threadIdx.x & 31;
    const int half = lane >> 4;
    const int r    = lane & 15;
    FragU u;
    const unsigned short* p = base + (size_t)r * ld + 8 * half;
    u.q[0] = *reinterpret_cast<const U128*>(p);
    u.q[1] = *reinterpret_cast<const U128*>(p + 16);
    return u.v;
}

// ---------- cast / transpose kernels ----------
__global__ void cast_bf16_kernel(const float* __restrict__ in, unsigned short* __restrict__ out, long n) {
    long i = (long)blockIdx.x * blockDim.x + threadIdx.x;
    if (i < n) out[i] = f2bf(in[i]);
}

// in: [K][N] f32 row-major  ->  out: [N][K] bf16 row-major
__global__ void transpose_bf16_kernel(const float* __restrict__ in, unsigned short* __restrict__ out,
                                      int K, int N) {
    long i = (long)blockIdx.x * blockDim.x + threadIdx.x;
    long total = (long)K * N;
    if (i < total) {
        int k = (int)(i / N);
        int n = (int)(i % N);
        out[(size_t)n * K + k] = f2bf(in[i]);
    }
}

// ---------- tiled WMMA GEMM: out[M][N] = A[M][K] * Bt[N][K]^T ----------
// block = 128 threads (4 waves); block tile 128x64; wave tile 32x64.
// B tile (64x32 bf16 = 4KB) double-buffered in LDS via async copies:
// issue stage(i+1) after the barrier, compute step i while DMA is in flight.
// MODE 0: bf16 row-major out. MODE 1: f32 row-major out.
// MODE 2: bf16 out scattered as Vt[b][g][d][t]  (N=512, T=2048 layout)
template <int MODE>
__global__ __launch_bounds__(128, 1) void gemm_bf16_kernel(const unsigned short* __restrict__ A,
                                                           const unsigned short* __restrict__ Bt,
                                                           void* __restrict__ outp,
                                                           int M, int N, int K) {
    __shared__ __align__(16) unsigned short Bs[2][64 * 32];   // two [n][k] tiles, 8KB

    const int wave = threadIdx.x >> 5;
    const int lane = threadIdx.x & 31;
    const int m0 = blockIdx.y * 128 + wave * 32;              // two 16-row tiles
    const int n0 = blockIdx.x * 64;

    auto stage = [&](int buf, int k) {
#pragma unroll
        for (int c = 0; c < 2; ++c) {
            const int q   = threadIdx.x + 128 * c;            // 0..255
            const int row = q >> 2;                           // 0..63
            const int kc  = (q & 3) * 8;                      // 0,8,16,24
            const unsigned short* gp = Bt + (size_t)(n0 + row) * K + k + kc;
            unsigned short* lp = &Bs[buf][row * 32 + kc];
#if USE_ASYNC_LDS
            __builtin_amdgcn_global_load_async_to_lds_b128((gv4i_p)gp, (lv4i_p)lp, 0, 0);
#else
            *reinterpret_cast<U128*>(lp) = *reinterpret_cast<const U128*>(gp);
#endif
        }
    };

    v8f acc[2][4] = { { {}, {}, {}, {} }, { {}, {}, {}, {} } };

    const int nk = K >> 5;
    stage(0, 0);
    for (int i = 0; i < nk; ++i) {
        const int k = i << 5;
#if USE_ASYNC_LDS
        asm volatile("s_wait_asynccnt 0x0" ::: "memory");     // this wave's copies done
#endif
        __syncthreads();                                      // all waves' copies visible;
                                                              // all waves done reading other buffer
        if (i + 1 < nk) stage((i + 1) & 1, k + 32);           // prefetch next tile (overlaps compute)

        const unsigned short* Bcur = Bs[i & 1];
        v16bf a0 = load_frag(A + (size_t)m0 * K + k, K);
        v16bf a1 = load_frag(A + (size_t)(m0 + 16) * K + k, K);
#pragma unroll
        for (int j = 0; j < 4; ++j) {
            v16bf b = load_frag(Bcur + 16 * j * 32, 32);
            acc[0][j] = __builtin_amdgcn_wmma_f32_16x16x32_bf16(
                false, a0, false, b, (short)0, acc[0][j], false, false);
            acc[1][j] = __builtin_amdgcn_wmma_f32_16x16x32_bf16(
                false, a1, false, b, (short)0, acc[1][j], false, false);
        }
    }

    const int col  = lane & 15;
    const int half = lane >> 4;
#pragma unroll
    for (int t = 0; t < 2; ++t) {
#pragma unroll
        for (int j = 0; j < 4; ++j) {
#pragma unroll
            for (int i = 0; i < 8; ++i) {
                int m = m0 + 16 * t + i + 8 * half;
                int n = n0 + 16 * j + col;
                float v = acc[t][j][i];
                if (MODE == 0) {
                    ((unsigned short*)outp)[(size_t)m * N + n] = f2bf(v);
                } else if (MODE == 1) {
                    ((float*)outp)[(size_t)m * N + n] = v;
                } else {
                    // Vt[b][g][d][t]; m = b*T + t, n = g*128 + d
                    int bidx = m >> 11, tt = m & 2047;
                    int g = n >> 7, d = n & 127;
                    ((unsigned short*)outp)[(((size_t)bidx * KV_HEADS + g) * HEAD_DIM + d) * SEQ_T + tt] = f2bf(v);
                }
            }
        }
    }
}

// ---------- flash attention (one wave = one 16-row query tile of one head) ----------
__global__ __launch_bounds__(32, 1) void attn_kernel(
        const unsigned short* __restrict__ Qb,  // [B*T][2048] bf16
        const unsigned short* __restrict__ Kb,  // [B*T][512]  bf16
        const unsigned short* __restrict__ Vt,  // [B][G][128][T] bf16
        const float* __restrict__ wlog,         // [G]
        unsigned short* __restrict__ Yb) {      // [B*T][2048] bf16
    __shared__ __align__(16) unsigned short Pbuf[16 * 32];

    const int qt = blockIdx.x;          // 0..T/16-1
    const int h  = blockIdx.y;          // 0..15
    const int b  = blockIdx.z;          // 0..1
    const int g  = h >> 2;              // group (H/G = 4)
    const int q0 = qt * 16;
    const int lane = threadIdx.x & 31;
    const int col  = lane & 15;
    const int half = lane >> 4;

    const float scale = wlog[g] * 0.08838834764831845f;  // 1/sqrt(128)

    // Q tile 16x128 as 4 A-fragments (kept resident: launch_bounds gives full VGPR budget)
    v16bf qf[4];
    const unsigned short* qbase = Qb + ((size_t)(b * SEQ_T + q0)) * D_MODEL + h * HEAD_DIM;
#pragma unroll
    for (int c = 0; c < 4; ++c) qf[c] = load_frag(qbase + 32 * c, D_MODEL);

    v8f o[8] = { {}, {}, {}, {}, {}, {}, {}, {} };
    float m_i[8], l_i[8];
#pragma unroll
    for (int i = 0; i < 8; ++i) { m_i[i] = -1e38f; l_i[i] = 0.0f; }

    const int nb = (q0 + 16 + 31) >> 5;           // causal: key blocks of 32
    for (int kb32 = 0; kb32 < nb; ++kb32) {
        const int kb = kb32 * 32;

        // S = Q K^T  (16 x 32 as two 16x16 tiles)
        v8f s0 = {}, s1 = {};
#pragma unroll
        for (int c = 0; c < 4; ++c) {
            v16bf k0 = load_frag(Kb + ((size_t)(b * SEQ_T + kb)) * KV_DIM + g * HEAD_DIM + 32 * c, KV_DIM);
            s0 = __builtin_amdgcn_wmma_f32_16x16x32_bf16(false, qf[c], false, k0, (short)0, s0, false, false);
            v16bf k1 = load_frag(Kb + ((size_t)(b * SEQ_T + kb + 16)) * KV_DIM + g * HEAD_DIM + 32 * c, KV_DIM);
            s1 = __builtin_amdgcn_wmma_f32_16x16x32_bf16(false, qf[c], false, k1, (short)0, s1, false, false);
        }

        // online softmax update; P written straight to LDS to keep live ranges short
        float alpha[8];
#pragma unroll
        for (int i = 0; i < 8; ++i) {
            const int row = q0 + i + 8 * half;
            float v0 = s0[i] * scale; if (kb + col      > row) v0 = -1e30f;
            float v1 = s1[i] * scale; if (kb + 16 + col > row) v1 = -1e30f;
            float rm = fmaxf(v0, v1);
#pragma unroll
            for (int off = 1; off < 16; off <<= 1) rm = fmaxf(rm, __shfl_xor(rm, off, 32));
            const float mn = fmaxf(m_i[i], rm);
            const float a  = __expf(m_i[i] - mn);
            const float e0 = __expf(v0 - mn);
            const float e1 = __expf(v1 - mn);
            float rs = e0 + e1;
#pragma unroll
            for (int off = 1; off < 16; off <<= 1) rs += __shfl_xor(rs, off, 32);
            l_i[i] = l_i[i] * a + rs;
            m_i[i] = mn;
            alpha[i] = a;
            Pbuf[(i + 8 * half) * 32 + col]      = f2bf(e0);
            Pbuf[(i + 8 * half) * 32 + 16 + col] = f2bf(e1);
        }
#pragma unroll
        for (int j = 0; j < 8; ++j)
#pragma unroll
            for (int i = 0; i < 8; ++i) o[j][i] = o[j][i] * alpha[i];

        __syncthreads();
        v16bf pf = load_frag(Pbuf, 32);

        // O += P * V  (V read from transposed layout Vt[d][t])
#pragma unroll
        for (int j = 0; j < 8; ++j) {
            v16bf vf = load_frag(Vt + (((size_t)(b * KV_HEADS + g)) * HEAD_DIM + 16 * j) * SEQ_T + kb, SEQ_T);
            o[j] = __builtin_amdgcn_wmma_f32_16x16x32_bf16(false, pf, false, vf, (short)0, o[j], false, false);
        }
        __syncthreads();
    }

    // epilogue: normalize and store into Y (bf16, [b*T + t][h*128 + d])
#pragma unroll
    for (int i = 0; i < 8; ++i) {
        const float inv = 1.0f / l_i[i];
        const int trow = q0 + i + 8 * half;
        const size_t rowbase = ((size_t)(b * SEQ_T + trow)) * D_MODEL + h * HEAD_DIM;
#pragma unroll
        for (int j = 0; j < 8; ++j)
            Yb[rowbase + 16 * j + col] = f2bf(o[j][i] * inv);
    }
}

// ---------- host launcher ----------
extern "C" void kernel_launch(void* const* d_in, const int* in_sizes, int n_in,
                              void* d_out, int out_size, void* d_ws, size_t ws_size,
                              hipStream_t stream) {
    const float* x    = (const float*)d_in[0];
    // d_in[1] attn_mask: causal mask applied analytically
    const float* Wq   = (const float*)d_in[2];
    const float* Wk   = (const float*)d_in[3];
    const float* Wv   = (const float*)d_in[4];
    const float* Wo   = (const float*)d_in[5];
    const float* wlog = (const float*)d_in[6];
    // d_in[7] weight_values: unused by the reference
    float* out = (float*)d_out;

    char* ws = (char*)d_ws;
    size_t off = 0;
    auto alloc = [&](size_t bytes) -> void* {
        void* p = (void*)(ws + off);
        off += (bytes + 255) & ~(size_t)255;
        return p;
    };

    unsigned short* xb  = (unsigned short*)alloc((size_t)M_ROWS * D_MODEL * 2);
    unsigned short* Qb  = (unsigned short*)alloc((size_t)M_ROWS * D_MODEL * 2);
    unsigned short* Kb  = (unsigned short*)alloc((size_t)M_ROWS * KV_DIM * 2);
    unsigned short* VtB = (unsigned short*)alloc((size_t)M_ROWS * KV_DIM * 2);
    unsigned short* Yb  = (unsigned short*)alloc((size_t)M_ROWS * D_MODEL * 2);
    unsigned short* Wqt = (unsigned short*)alloc((size_t)D_MODEL * D_MODEL * 2);
    unsigned short* Wkt = (unsigned short*)alloc((size_t)D_MODEL * KV_DIM * 2);
    unsigned short* Wvt = (unsigned short*)alloc((size_t)D_MODEL * KV_DIM * 2);
    unsigned short* Wot = (unsigned short*)alloc((size_t)D_MODEL * D_MODEL * 2);
    (void)ws_size; (void)n_in; (void)in_sizes; (void)out_size;

    // 1) precision conversion + weight transposes
    {
        long n = (long)M_ROWS * D_MODEL;
        cast_bf16_kernel<<<(int)((n + 255) / 256), 256, 0, stream>>>(x, xb, n);
        long nq = (long)D_MODEL * D_MODEL;
        transpose_bf16_kernel<<<(int)((nq + 255) / 256), 256, 0, stream>>>(Wq, Wqt, D_MODEL, D_MODEL);
        transpose_bf16_kernel<<<(int)((nq + 255) / 256), 256, 0, stream>>>(Wo, Wot, D_MODEL, D_MODEL);
        long nk = (long)D_MODEL * KV_DIM;
        transpose_bf16_kernel<<<(int)((nk + 255) / 256), 256, 0, stream>>>(Wk, Wkt, D_MODEL, KV_DIM);
        transpose_bf16_kernel<<<(int)((nk + 255) / 256), 256, 0, stream>>>(Wv, Wvt, D_MODEL, KV_DIM);
    }

    // 2) projections
    {
        dim3 gq(D_MODEL / 64, M_ROWS / 128);
        gemm_bf16_kernel<0><<<gq, 128, 0, stream>>>(xb, Wqt, Qb, M_ROWS, D_MODEL, D_MODEL);
        dim3 gk(KV_DIM / 64, M_ROWS / 128);
        gemm_bf16_kernel<0><<<gk, 128, 0, stream>>>(xb, Wkt, Kb, M_ROWS, KV_DIM, D_MODEL);
        gemm_bf16_kernel<2><<<gk, 128, 0, stream>>>(xb, Wvt, VtB, M_ROWS, KV_DIM, D_MODEL);
    }

    // 3) grouped causal attention
    {
        dim3 ga(SEQ_T / 16, N_HEADS, BATCH);
        attn_kernel<<<ga, 32, 0, stream>>>(Qb, Kb, VtB, wlog, Yb);
    }

    // 4) output projection (f32 out)
    {
        dim3 go(D_MODEL / 64, M_ROWS / 128);
        gemm_bf16_kernel<1><<<go, 128, 0, stream>>>(Yb, Wot, out, M_ROWS, D_MODEL, D_MODEL);
    }
}
